// KnowEncoder_49478023250052
// MI455X (gfx1250) — matrix operand order repaired and checked by reading, not compiled, not run on hardware
//
#include <hip/hip_runtime.h>

#define L_    16
#define H_    16
#define D_    64
#define GK    128    // G
#define COUT  256    // Cout_g
#define CIN   192    // Cin_g
#define NSEQ  256    // n
#define EDIM  768    // E
#define BK    16     // B*K
#define BB    4      // B
#define KKK   4      // k

#define WSTR  200    // padded LDS row stride (halfs): 400B = 100 DWORDs -> 16 banks, 16B aligned
#define XSTR  200

typedef __attribute__((ext_vector_type(16))) _Float16 v16h;
typedef __attribute__((ext_vector_type(8)))  _Float16 v8h;
typedef __attribute__((ext_vector_type(8)))  float    v8f;

// ---------------------------------------------------------------------------
// Kernel 1: per-(b,j) sequence lengths, prefix offsets, per-b totals -> ws
// ws[0..15]=lens, ws[16..31]=offs, ws[32..35]=totals
// ---------------------------------------------------------------------------
__global__ void lens_kernel(const int* __restrict__ mask, int* __restrict__ ws) {
    __shared__ int sums[BK];
    int tid = threadIdx.x;
    if (tid < BK) sums[tid] = 0;
    __syncthreads();
    int row  = tid >> 4;
    int col0 = tid & 15;
    int part = 0;
    for (int t = col0; t < NSEQ; t += 16) part += mask[row * NSEQ + t];
    atomicAdd(&sums[row], part);
    __syncthreads();
    if (tid == 0) {
        for (int i = 0; i < BB; ++i) {
            int acc = 0;
            for (int j = 0; j < KKK; ++j) {
                int l = sums[i * KKK + j];
                ws[i * KKK + j]      = l;    // lens
                ws[BK + i * KKK + j] = acc;  // offsets
                acc += l;
            }
            ws[2 * BK + i] = acc;            // totals
        }
    }
}

// ---------------------------------------------------------------------------
// Kernel 2: masked-mean pooled output (16, 768)
// ---------------------------------------------------------------------------
__global__ void pooled_kernel(const float* __restrict__ lh,
                              const int* __restrict__ mask,
                              const int* __restrict__ ws,
                              float* __restrict__ pooled) {
    int bk  = blockIdx.x;
    int tid = threadIdx.x;
    float inv = 1.0f / (float)ws[bk];
    const float* base = lh + (size_t)bk * NSEQ * EDIM;
    const int*   m    = mask + bk * NSEQ;
    for (int c = tid; c < EDIM; c += blockDim.x) {
        float s = 0.0f;
        for (int t = 0; t < NSEQ; ++t)
            if (m[t]) s += base[(size_t)t * EDIM + c];
        pooled[bk * EDIM + c] = s * inv;
    }
}

// ---------------------------------------------------------------------------
// Kernel 3: valid mask (B, Tmax) int32
// ---------------------------------------------------------------------------
__global__ void valid_kernel(const int* __restrict__ ws, int* __restrict__ valid, int Tmax) {
    int idx = blockIdx.x * blockDim.x + threadIdx.x;
    if (idx >= BB * Tmax) return;
    int b = idx / Tmax;
    int T = idx - b * Tmax;
    valid[idx] = (T < ws[2 * BK + b]) ? 1 : 0;
}

// ---------------------------------------------------------------------------
// Kernel 4: block-diagonal GEMM with WMMA f32<-f16, fused gather/scatter.
// One workgroup (8 wave32) per (bk, g). Each wave owns 2 full M-tile rows:
// A fragments loaded once per row (16x reuse), 2 concurrent accumulators for
// XDL pipelining, 32-bit offset stores from the uniform batch base.
// ---------------------------------------------------------------------------
__global__ __launch_bounds__(256) void gemm_kernel(const float* __restrict__ lh,
                                                   const float* __restrict__ weight,
                                                   const float* __restrict__ bias,
                                                   const int* __restrict__ ws,
                                                   float* __restrict__ batch,
                                                   int Tmax) {
    extern __shared__ _Float16 smem[];
    _Float16* Wl = smem;                       // [COUT][WSTR]  W_g as f16
    _Float16* Xl = smem + COUT * WSTR;         // [NSEQ][XSTR]  X_g^T: [t][i]

    const int bk  = blockIdx.x >> 7;           // / GK
    const int g   = blockIdx.x & (GK - 1);
    const int tid = threadIdx.x;
    const int start = (g & 3) * CIN;           // (g*CIN) % EDIM, never wraps

    // gfx1250 prefetch path (global_prefetch_b8)
    __builtin_prefetch(bias + g * COUT, 0, 3);

    // ---- stage W_g (256x192 f32 -> f16) ----
    const float* wsrc = weight + (size_t)g * COUT * CIN;
    #pragma unroll 4
    for (int it = 0; it < 48; ++it) {
        int q  = it * 256 + tid;
        int o  = q / 48;              // 48 float4 per row of 192
        int kf = (q - o * 48) * 4;
        float4 f = *(const float4*)(wsrc + (size_t)o * CIN + kf);
        _Float16* d = Wl + o * WSTR + kf;
        d[0] = (_Float16)f.x; d[1] = (_Float16)f.y;
        d[2] = (_Float16)f.z; d[3] = (_Float16)f.w;
    }
    // ---- stage X_g^T (256 tokens x 192 ch, f32 -> f16) ----
    const float* xsrc = lh + (size_t)bk * NSEQ * EDIM + start;
    #pragma unroll 4
    for (int it = 0; it < 48; ++it) {
        int q  = it * 256 + tid;
        int t  = q / 48;
        int kf = (q - t * 48) * 4;
        float4 f = *(const float4*)(xsrc + (size_t)t * EDIM + kf);
        _Float16* d = Xl + t * XSTR + kf;
        d[0] = (_Float16)f.x; d[1] = (_Float16)f.y;
        d[2] = (_Float16)f.z; d[3] = (_Float16)f.w;
    }

    const int wave = tid >> 5;
    const int lane = tid & 31;
    const int hi   = lane >> 4;       // 0/1 half-wave
    const int lo   = lane & 15;

    const int len = ws[bk];
    const int off = ws[BK + bk];
    const int b   = bk >> 2;          // / k

    // Output decomposition for this workgroup: l*2+s == g>>2 (uniform),
    // h = hi*8 + v, d = (g*16 + tm) & 63, T = off + token.
    const unsigned hstride = (unsigned)Tmax * D_;                       // elems per h step
    const unsigned rowbase = ((((unsigned)(g >> 2) * BB + (unsigned)b) * H_)
                              + (unsigned)hi * 8) * hstride;

    __syncthreads();

    #pragma unroll
    for (int tmi = 0; tmi < 2; ++tmi) {
        const int tm   = wave * 2 + tmi;
        const int orow = tm * 16 + lo;
        const _Float16* arow = Wl + orow * WSTR + hi * 16;

        // A fragments for the whole K (6 x v16h), reused across 16 N-tiles.
        // A in contiguous k order (same as B): halfs 0-15 -> K=kc*32+16*hi+0..15
        v16h afrag[6];
        #pragma unroll
        for (int kc = 0; kc < 6; ++kc) {
            union { v16h v; v8h h[2]; } ua;
            ua.h[0] = *(const v8h*)(arow + kc * 32);
            ua.h[1] = *(const v8h*)(arow + kc * 32 + 8);
            afrag[kc] = ua.v;
        }

        const int obase = tm * 16 + hi * 8;    // C/D layout: VGPR v -> M = obase+v
        const float* bsrc = bias + g * COUT + obase;
        float bias8[8];
        #pragma unroll
        for (int v = 0; v < 8; ++v) bias8[v] = bsrc[v];

        const unsigned dd       = (unsigned)((g * 16 + tm) & 63);
        const unsigned tilebase = rowbase + dd;

        for (int tn = 0; tn < 16; tn += 2) {
            const int tc0 = tn * 16 + lo;      // B col N (token), tile tn
            const int tc1 = tc0 + 16;          // tile tn+1
            const _Float16* brow0 = Xl + tc0 * XSTR + hi * 16;
            const _Float16* brow1 = Xl + tc1 * XSTR + hi * 16;

            v8f acc0, acc1;
            #pragma unroll
            for (int v = 0; v < 8; ++v) { acc0[v] = bias8[v]; acc1[v] = bias8[v]; }

            #pragma unroll
            for (int kc = 0; kc < 6; ++kc) {
                // B in contiguous k order (same as A): halfs 0-15 -> K = kc*32+16*hi+0..15
                union { v16h v; v8h h[2]; } ub0, ub1;
                ub0.h[0] = *(const v8h*)(brow0 + kc * 32);
                ub0.h[1] = *(const v8h*)(brow0 + kc * 32 + 8);
                ub1.h[0] = *(const v8h*)(brow1 + kc * 32);
                ub1.h[1] = *(const v8h*)(brow1 + kc * 32 + 8);
                acc0 = __builtin_amdgcn_wmma_f32_16x16x32_f16(
                    false, afrag[kc], false, ub0.v, (short)0, acc0, false, false);
                acc1 = __builtin_amdgcn_wmma_f32_16x16x32_f16(
                    false, afrag[kc], false, ub1.v, (short)0, acc1, false, false);
            }

            // Scatter with 32-bit element offsets from uniform batch base.
            if (tc0 < len) {
                unsigned a0 = tilebase + (unsigned)(off + tc0) * D_;
                #pragma unroll
                for (int v = 0; v < 8; ++v)
                    batch[a0 + (unsigned)v * hstride] = acc0[v];
            }
            if (tc1 < len) {
                unsigned a1 = tilebase + (unsigned)(off + tc1) * D_;
                #pragma unroll
                for (int v = 0; v < 8; ++v)
                    batch[a1 + (unsigned)v * hstride] = acc1[v];
            }
        }
    }
}

// ---------------------------------------------------------------------------
extern "C" void kernel_launch(void* const* d_in, const int* in_sizes, int n_in,
                              void* d_out, int out_size, void* d_ws, size_t ws_size,
                              hipStream_t stream) {
    const float* lh     = (const float*)d_in[0];
    const int*   mask   = (const int*)d_in[1];
    const float* weight = (const float*)d_in[2];
    const float* bias   = (const float*)d_in[3];
    // d_in[4] = k (== 4), baked into constants

    float* pooled = (float*)d_out;
    float* batch  = pooled + BK * EDIM;

    // out_size = 12288 + (131072 + 4) * Tmax
    const int per_T = L_ * 2 * BB * H_ * D_ + BB;      // 131076
    int Tmax = (out_size - BK * EDIM) / per_T;
    if (Tmax < 1) Tmax = 1;

    size_t batch_elems = (size_t)L_ * 2 * BB * H_ * (size_t)Tmax * D_;
    int* valid = (int*)(batch + batch_elems);
    int* ws    = (int*)d_ws;

    lens_kernel<<<1, 256, 0, stream>>>(mask, ws);
    pooled_kernel<<<BK, 256, 0, stream>>>(lh, mask, ws, pooled);
    hipMemsetAsync(batch, 0, batch_elems * sizeof(float), stream);
    valid_kernel<<<(BB * Tmax + 255) / 256, 256, 0, stream>>>(ws, valid, Tmax);

    const size_t lds_bytes = (size_t)(COUT * WSTR + NSEQ * XSTR) * sizeof(_Float16); // 200 KB
    gemm_kernel<<<BK * GK, 256, lds_bytes, stream>>>(lh, weight, bias, ws, batch, Tmax);
}
